// CNN3_50998441672722
// MI455X (gfx1250) — matrix operand order repaired and checked by reading, not compiled, not run on hardware
//
#include <hip/hip_runtime.h>
#include <math.h>

#define NB    8
#define NT    1023
#define NROWS (NB*NT)        // 8184
#define NFFT  512
#define HOP   256
#define NF    257
#define NCOL  514            // 2 * 257 (re/im interleaved)
#define NTN   33             // ceil(514/16)
#define NPADC (NTN*16)       // 528
#define SLEN  262144
#define FP    276            // padded freq rows for LDS tiles: f in [-2, 273] -> idx f+2
#define HP    ((size_t)NROWS * NF)   // one channel plane of H

typedef __attribute__((ext_vector_type(16))) _Float16 v16h;
typedef __attribute__((ext_vector_type(8)))  float    v8f;

union V16H { _Float16 h[16]; v16h v; };
union V8F  { float    f[8];  v8f  v; };

__device__ __forceinline__ v8f wmma_f16(v16h a, v16h b, v8f c) {
  // D = A(16x32 f16) * B(32x16 f16) + C(16x16 f32)
  return __builtin_amdgcn_wmma_f32_16x16x32_f16(false, a, false, b, (short)0, c, false, false);
}

// ---------------------------------------------------------------------------
// Windowed rfft basis, stored directly in WMMA B-fragment order:
// element index = ((nt*16 + ks)*32 + lane)*16 + j ; col n = nt*16+(lane&15),
// k (time sample) = ks*32 + j + 16*(lane>>4).  B[k][col] = win(k)*{cos,-sin}.
// ---------------------------------------------------------------------------
__global__ void k_basis(_Float16* __restrict__ bf) {
  int blk  = blockIdx.x;          // nt*16 + ks
  int nt   = blk >> 4, ks = blk & 15;
  int lane = threadIdx.x;         // 0..31
  int hi   = lane >> 4;
  int col  = nt*16 + (lane & 15);
  V16H o;
#pragma unroll
  for (int j = 0; j < 16; ++j) {
    int k = ks*32 + j + 16*hi;    // time sample n in [0,512)
    float v = 0.f;
    if (col < NCOL) {
      int f = col >> 1;
      float win = 0.5f - 0.5f * __cosf((float)(2.0 * M_PI / NFFT) * (float)k);
      int ph = (k * f) & (NFFT - 1);         // exact phase reduction
      float s, c;
      __sincosf((float)(2.0 * M_PI / NFFT) * (float)ph, &s, &c);
      v = win * ((col & 1) ? -s : c);
    }
    o.h[j] = (_Float16)v;
  }
  *(v16h*)&bf[(size_t)(blk*32 + lane) * 16] = o.v;
}

// ---------------------------------------------------------------------------
// STFT as GEMM:  spec[8184 x 514] = frames[8184 x 512] * basis[512 x 514]
// One block = one 16-row M tile (frames staged as f16 in LDS); 4 waves split
// the 33 N tiles; K = 512 -> 16 WMMA k-steps; B fragments double-buffered.
// ---------------------------------------------------------------------------
__global__ __launch_bounds__(128) void k_stft(const float* __restrict__ x,
                                              const _Float16* __restrict__ bf,
                                              float* __restrict__ spec) {
  __shared__ _Float16 Alds[16][NFFT];       // 16 KB
  int mt = blockIdx.x;
  int tid = threadIdx.x, lane = tid & 31, wave = tid >> 5;
  // warm up L2 for this wave's first B-fragment block
  __builtin_prefetch(&bf[(size_t)(wave*16)*32*16 + (size_t)lane*16], 0, 1);
  for (int idx = tid; idx < 16*NFFT; idx += blockDim.x) {
    int m = idx >> 9, n = idx & (NFFT-1);
    int row = mt*16 + m;
    float v = 0.f;
    if (row < NROWS) {
      int b = row / NT, t = row - b*NT;
      v = x[(size_t)b*SLEN + (size_t)t*HOP + n];
    }
    Alds[m][n] = (_Float16)v;
  }
  __syncthreads();
  int hi = lane >> 4, mlo = lane & 15;
  for (int nt = wave; nt < NTN; nt += 4) {   // wave-uniform trip count
    v8f acc = {0.f,0.f,0.f,0.f,0.f,0.f,0.f,0.f};
    const _Float16* bp = &bf[(size_t)(nt*16)*32*16 + (size_t)lane*16];
    v16h bcur = *(const v16h*)bp;            // stage 0
#pragma unroll
    for (int ks = 0; ks < 16; ++ks) {
      v16h bnxt = bcur;
      if (ks < 15) bnxt = *(const v16h*)(bp + (size_t)(ks+1)*32*16); // prefetch next
      V16H a;
#pragma unroll
      for (int j = 0; j < 16; ++j) {
        int k = ks*32 + j + 16*hi;   // contiguous k layout (matches B)
        a.h[j] = Alds[mlo][k];
      }
      acc = wmma_f16(a.v, bcur, acc);
      bcur = bnxt;
    }
    V8F d; d.v = acc;
    int col = nt*16 + mlo;
#pragma unroll
    for (int r = 0; r < 8; ++r) {
      int row = mt*16 + r + 8*hi;            // D layout: m = r + 8*(lane>>4)
      if (row < NROWS) spec[(size_t)row*NPADC + col] = d.f[r];
    }
  }
}

// ---------------------------------------------------------------------------
// l0: TD block on spectrum, 2 -> relu(4) -> 8, relu.  One block per (b,t).
// H is channel-plane-major: H[c*HP + row*NF + f]  (dense per-channel streams)
// ---------------------------------------------------------------------------
__global__ __launch_bounds__(128) void k_l0(const float* __restrict__ spec,
    const float* __restrict__ W1, const float* __restrict__ b1,
    const float* __restrict__ W2, const float* __restrict__ b2,
    float* __restrict__ H) {
  __shared__ float srow[NCOL];
  __shared__ float h1[NF + 2][4];            // shifted by +1, zero halo
  int row = blockIdx.x, tid = threadIdx.x;
  for (int c = tid; c < NCOL; c += blockDim.x) srow[c] = spec[(size_t)row*NPADC + c];
  if (tid < 4) { h1[0][tid] = 0.f; h1[NF+1][tid] = 0.f; }
  __syncthreads();
  for (int f = tid; f < NF; f += blockDim.x) {
#pragma unroll
    for (int o = 0; o < 4; ++o) {
      float a = b1[o];
#pragma unroll
      for (int i = 0; i < 3; ++i) {
        int fs = f + i - 1;
        if (fs >= 0 && fs < NF) {
          a += srow[2*fs]   * W1[(i*2+0)*4 + o];
          a += srow[2*fs+1] * W1[(i*2+1)*4 + o];
        }
      }
      h1[f+1][o] = fmaxf(a, 0.f);
    }
  }
  __syncthreads();
  for (int f = tid; f < NF; f += blockDim.x) {
#pragma unroll
    for (int o = 0; o < 8; ++o) {
      float a = b2[o];
#pragma unroll
      for (int i = 0; i < 3; ++i)
#pragma unroll
        for (int c = 0; c < 4; ++c)
          a += h1[f+i][c] * W2[(i*4+c)*8 + o];
      H[(size_t)o*HP + (size_t)row*NF + f] = fmaxf(a, 0.f);
    }
  }
}

// ---------------------------------------------------------------------------
// Per-row LayerNorm statistics over (F, Cin): reads exactly Cin dense planes.
// ---------------------------------------------------------------------------
__global__ __launch_bounds__(256) void k_stats(const float* __restrict__ H, int C,
                                               float* __restrict__ mu, float* __restrict__ rv) {
  __shared__ float ssum[256], ssq[256];
  int row = blockIdx.x, tid = threadIdx.x;
  float s = 0.f, q = 0.f;
  int n = NF * C;
  for (int idx = tid; idx < n; idx += 256) {
    int c = idx / NF, f = idx - c*NF;        // f-contiguous per thread step
    float v = H[(size_t)c*HP + (size_t)row*NF + f];
    s += v; q += v*v;
  }
  ssum[tid] = s; ssq[tid] = q;
  __syncthreads();
  for (int st = 128; st > 0; st >>= 1) {
    if (tid < st) { ssum[tid] += ssum[tid+st]; ssq[tid] += ssq[tid+st]; }
    __syncthreads();
  }
  if (tid == 0) {
    float m   = ssum[0] / (float)n;
    float var = ssq[0] / (float)n - m*m;
    mu[row] = m;
    rv[row] = rsqrtf(var + 1e-5f);
  }
}

// ---------------------------------------------------------------------------
// One dilation layer: LN + causal depthwise(T,3) fused into f16 LDS tile,
// then conv1 (3*Cin -> 8) and conv2 (24 -> 8) over frequency via WMMA.
// Writes relu result into channel planes [co, co+8) of H (LDS-staged,
// plane-coalesced writeback).
// ---------------------------------------------------------------------------
__global__ __launch_bounds__(128) void k_layer(
    const float* __restrict__ H, const float* __restrict__ mu, const float* __restrict__ rv,
    const float* __restrict__ g, const float* __restrict__ beta, const float* __restrict__ dw,
    const float* __restrict__ W1, const float* __restrict__ b1,
    const float* __restrict__ W2, const float* __restrict__ b2,
    int Cin, int co, float* __restrict__ Hout) {
  __shared__ _Float16 yls[FP][24];           // normalized+dw tile, zero halo
  __shared__ _Float16 h1ls[FP][8];           // conv1 output, zero halo
  __shared__ float    outls[NF][8];          // conv2 output staging
  int row = blockIdx.x, tid = threadIdx.x, lane = tid & 31, wave = tid >> 5;
  int b = row / NT, t = row - b*NT;

  for (int idx = tid; idx < FP*24; idx += blockDim.x) (&yls[0][0])[idx]  = (_Float16)0.f;
  for (int idx = tid; idx < FP*8;  idx += blockDim.x) (&h1ls[0][0])[idx] = (_Float16)0.f;
  __syncthreads();

  // Phase 1: LayerNorm + causal depthwise over time (taps t-2, t-1, t)
  int nfc = NF * Cin;
  for (int idx = tid; idx < nfc; idx += blockDim.x) {
    int c = idx / NF, f = idx - c*NF;        // f-contiguous -> dense plane reads
    int fc = f*Cin + c;                      // parameter index (F, Cin) row-major
    float gg = g[fc], bb = beta[fc];
    float acc = 0.f;
#pragma unroll
    for (int i = 0; i < 3; ++i) {
      int tt = t - 2 + i;
      if (tt >= 0) {
        int r2 = b*NT + tt;
        float xv = H[(size_t)c*HP + (size_t)r2*NF + f];
        float xn = (xv - mu[r2]) * rv[r2] * gg + bb;
        acc += xn * dw[fc*3 + i];
      }
    }
    yls[f+2][c] = (_Float16)acc;
  }
  __syncthreads();

  int hi = lane >> 4, nlo = lane & 15, mlo = lane & 15;
  int Ktot = 3*Cin;
  int nks  = (Ktot + 31) >> 5;               // 1..3 K-steps

  // B fragments (weights) built once in registers (B layout: k = j + 16*hi)
  V16H bw[3];
  for (int ks = 0; ks < 3; ++ks) {
#pragma unroll
    for (int j = 0; j < 16; ++j) {
      int k = ks*32 + j + 16*hi;
      float v = 0.f;
      if (nlo < 8 && k < Ktot) {
        int i = k / Cin, c = k - i*Cin;
        v = W1[(i*Cin + c)*8 + nlo];
      }
      bw[ks].h[j] = (_Float16)v;
    }
  }
  V16H bw2;
#pragma unroll
  for (int j = 0; j < 16; ++j) {
    int k = j + 16*hi;
    float v = (nlo < 8 && k < 24) ? W2[k*8 + nlo] : 0.f;
    bw2.h[j] = (_Float16)v;
  }
  float b1v = (nlo < 8) ? b1[nlo] : 0.f;
  float b2v = (nlo < 8) ? b2[nlo] : 0.f;

  // Phase 2: conv1 via WMMA, 17 M tiles of 16 freq positions split over waves
  for (int mt = wave; mt < 17; mt += 4) {
    v8f acc = {0.f,0.f,0.f,0.f,0.f,0.f,0.f,0.f};
    for (int ks = 0; ks < nks; ++ks) {
      V16H a;
#pragma unroll
      for (int j = 0; j < 16; ++j) {
        int k = ks*32 + j + 16*hi;
        _Float16 v = (_Float16)0.f;
        if (k < Ktot) {
          int i = k / Cin, c = k - i*Cin;
          v = yls[mt*16 + mlo + i - 1 + 2][c];     // SAME pad via zero halo
        }
        a.h[j] = v;
      }
      acc = wmma_f16(a.v, bw[ks].v, acc);
    }
    V8F d; d.v = acc;
#pragma unroll
    for (int r = 0; r < 8; ++r) {
      int f = mt*16 + r + 8*hi;
      if (nlo < 8 && f < NF) h1ls[f+2][nlo] = (_Float16)fmaxf(d.f[r] + b1v, 0.f);
    }
  }
  __syncthreads();

  // Phase 3: conv2 via WMMA (K = 24 padded to 32), relu, stage in LDS
  for (int mt = wave; mt < 17; mt += 4) {
    V16H a;
#pragma unroll
    for (int j = 0; j < 16; ++j) {
      int k = j + 16*hi;
      _Float16 v = (_Float16)0.f;
      if (k < 24) {
        int i = k >> 3, c = k & 7;
        v = h1ls[mt*16 + mlo + i - 1 + 2][c];
      }
      a.h[j] = v;
    }
    v8f z = {0.f,0.f,0.f,0.f,0.f,0.f,0.f,0.f};
    v8f acc = wmma_f16(a.v, bw2.v, z);
    V8F d; d.v = acc;
#pragma unroll
    for (int r = 0; r < 8; ++r) {
      int f = mt*16 + r + 8*hi;
      if (nlo < 8 && f < NF) outls[f][nlo] = fmaxf(d.f[r] + b2v, 0.f);
    }
  }
  __syncthreads();

  // Plane-coalesced writeback of the 8 new channels
  for (int idx = tid; idx < 8*NF; idx += blockDim.x) {
    int c = idx / NF, f = idx - c*NF;
    Hout[(size_t)(co + c)*HP + (size_t)row*NF + f] = outls[f][c];
  }
}

// ---------------------------------------------------------------------------
// Head: fc1 (k=3 valid, 32->1, relu) then fc2 (255-dot) + sigmoid per (b,t).
// ---------------------------------------------------------------------------
__global__ __launch_bounds__(128) void k_head(const float* __restrict__ H,
    const float* __restrict__ fc1W, const float* __restrict__ fc1b,
    const float* __restrict__ fc2W, const float* __restrict__ fc2b,
    float* __restrict__ out) {
  __shared__ float hrow[NF][32];
  __shared__ float red[128];
  int row = blockIdx.x, tid = threadIdx.x;
  for (int idx = tid; idx < 32*NF; idx += blockDim.x) {
    int c = idx / NF, f = idx - c*NF;        // dense plane reads
    hrow[f][c] = H[(size_t)c*HP + (size_t)row*NF + f];
  }
  __syncthreads();
  float part = 0.f;
  for (int f = tid; f < NF-2; f += blockDim.x) {
    float a = fc1b[0];
#pragma unroll
    for (int i = 0; i < 3; ++i)
      for (int c = 0; c < 32; ++c)
        a += hrow[f+i][c] * fc1W[i*32 + c];
    part += fmaxf(a, 0.f) * fc2W[f];
  }
  red[tid] = part;
  __syncthreads();
  for (int st = 64; st > 0; st >>= 1) {
    if (tid < st) red[tid] += red[tid + st];
    __syncthreads();
  }
  if (tid == 0) {
    float v = red[0] + fc2b[0];
    out[row] = 1.f / (1.f + __expf(-v));
  }
}

// ---------------------------------------------------------------------------
extern "C" void kernel_launch(void* const* d_in, const int* in_sizes, int n_in,
                              void* d_out, int out_size, void* d_ws, size_t ws_size,
                              hipStream_t stream) {
  (void)in_sizes; (void)n_in; (void)out_size; (void)ws_size;
  const float* x    = (const float*)d_in[0];
  const float* l0W1 = (const float*)d_in[1];
  const float* l0b1 = (const float*)d_in[2];
  const float* l0W2 = (const float*)d_in[3];
  const float* l0b2 = (const float*)d_in[4];
  const float *Lg[3], *Lbn[3], *Ldw[3], *LW1[3], *Lb1[3], *LW2[3], *Lb2[3];
  for (int l = 0; l < 3; ++l) {
    const float* const* p = (const float* const*)(d_in + 5 + 7*l);
    Lg[l] = p[0]; Lbn[l] = p[1]; Ldw[l] = p[2];
    LW1[l] = p[3]; Lb1[l] = p[4]; LW2[l] = p[5]; Lb2[l] = p[6];
  }
  const float* fc1W = (const float*)d_in[26];
  const float* fc1b = (const float*)d_in[27];
  const float* fc2W = (const float*)d_in[28];
  const float* fc2b = (const float*)d_in[29];
  float* out = (float*)d_out;

  char* ws = (char*)d_ws;
  size_t off = 0;
  _Float16* bf = (_Float16*)(ws + off); off += (size_t)NTN*16*32*16*2;   // 0.54 MB
  float* spec  = (float*)(ws + off);    off += (size_t)8192*NPADC*4;     // 17.3 MB
  float* H     = (float*)(ws + off);    off += (size_t)32*HP*4;          // 269.2 MB
  float* mu    = (float*)(ws + off);    off += (size_t)8192*4;
  float* rv    = (float*)(ws + off);    off += (size_t)8192*4;

  k_basis<<<NTN*16, 32, 0, stream>>>(bf);
  k_stft <<<512, 128, 0, stream>>>(x, bf, spec);
  k_l0   <<<NROWS, 128, 0, stream>>>(spec, l0W1, l0b1, l0W2, l0b2, H);

  const int cins[3] = {8, 16, 24};
  for (int l = 0; l < 3; ++l) {
    k_stats<<<NROWS, 256, 0, stream>>>(H, cins[l], mu, rv);
    k_layer<<<NROWS, 128, 0, stream>>>(H, mu, rv, Lg[l], Lbn[l], Ldw[l],
                                       LW1[l], Lb1[l], LW2[l], Lb2[l],
                                       cins[l], cins[l], H);
  }
  k_head<<<NROWS, 128, 0, stream>>>(H, fc1W, fc1b, fc2W, fc2b, out);
}
